// self_attention_heads_1374389534913
// MI455X (gfx1250) — matrix operand order repaired and checked by reading, not compiled, not run on hardware
//
#include <hip/hip_runtime.h>
#include <hip/hip_bf16.h>

#define DEV __device__ __forceinline__

typedef __attribute__((ext_vector_type(16))) __bf16 v16bf;
typedef __attribute__((ext_vector_type(8)))  __bf16 v8bf;
typedef __attribute__((ext_vector_type(8)))  float  v8f;

static constexpr int BB = 4;     // batch
static constexpr int LL = 2048;  // sequence
static constexpr int DD = 64;    // feature
static constexpr int HH = 8;     // heads

// ---------------------------------------------------------------------------
// WMMA helpers (CDNA5 v_wmma_f32_16x16x32_bf16)
// ---------------------------------------------------------------------------
DEV v8f wmma_bf(v16bf a, v16bf b, v8f c) {
  // (neg_a, A, neg_b, B, c_mod, C, reuse_a, reuse_b)
  return __builtin_amdgcn_wmma_f32_16x16x32_bf16(false, a, false, b, (short)0, c, false, false);
}

// A fragment: 16(M) x 32(K) bf16, row-major source, ld in elements.
// lanes 0-15 : row = row_lane, K = k0+0..7 and k0+16..23
// lanes 16-31: row = row_lane, K = k0+8..15 and k0+24..31
DEV v16bf frag_a(const __bf16* base, int ld, int row_lane, int k0, int lane) {
  const __bf16* p = base + (size_t)row_lane * ld + k0 + ((lane & 16) ? 8 : 0);
  v8bf lo = *(const v8bf*)p;
  v8bf hi = *(const v8bf*)(p + 16);
  return __builtin_shufflevector(lo, hi, 0,1,2,3,4,5,6,7,8,9,10,11,12,13,14,15);
}

// B fragment: 32(K) x 16(N), sourced from B^T stored row-major (row = N, ld elems).
// lanes 0-15 : col = col_lane, K = k0+0..7 and k0+16..23
// lanes 16-31: col = col_lane, K = k0+8..15 and k0+24..31
DEV v16bf frag_b(const __bf16* bt, int ld, int col_lane, int k0, int lane) {
  const __bf16* p = bt + (size_t)col_lane * ld + k0 + ((lane & 16) ? 8 : 0);
  v8bf lo = *(const v8bf*)p;
  v8bf hi = *(const v8bf*)(p + 16);
  return __builtin_shufflevector(lo, hi, 0,1,2,3,4,5,6,7,8,9,10,11,12,13,14,15);
}

// ---------------------------------------------------------------------------
// prep kernels
// ---------------------------------------------------------------------------
__global__ __launch_bounds__(256) void k_f32_to_bf16(const float* __restrict__ s,
                                                     __bf16* __restrict__ d, int n) {
  for (int i = blockIdx.x * blockDim.x + threadIdx.x; i < n; i += gridDim.x * blockDim.x)
    d[i] = (__bf16)s[i];
}

// x (B,L,D) f32 -> xt (B,D,L) bf16
__global__ __launch_bounds__(256) void k_transpose(const float* __restrict__ x,
                                                   __bf16* __restrict__ xt) {
  const int n = BB * LL * DD;
  for (int i = blockIdx.x * blockDim.x + threadIdx.x; i < n; i += gridDim.x * blockDim.x) {
    int b = i / (LL * DD);
    int r = i - b * (LL * DD);
    int l = r / DD, d = r - l * DD;
    xt[((size_t)b * DD + d) * LL + l] = (__bf16)x[i];
  }
}

// ---------------------------------------------------------------------------
// GEMM1: sim[b,h] = kernel_h (LxL) @ x_b (LxD), output bf16 row-major (B,H,L,D)
// grid (8, H, 2*B), block 128 (4 waves); each wave = one 64x64 output tile
// ---------------------------------------------------------------------------
__global__ __launch_bounds__(128, 1)
void k_gemm1(const __bf16* __restrict__ kbf,
             const __bf16* __restrict__ x0t,
             const __bf16* __restrict__ x1t,
             __bf16* __restrict__ sim1,
             __bf16* __restrict__ sim2) {
  const int lane = threadIdx.x & 31;
  const int wave = threadIdx.x >> 5;
  const int lt   = blockIdx.x * 4 + wave;      // 0..31  (M tile of 64)
  const int h    = blockIdx.y;
  const int z    = blockIdx.z;
  const int which = z >> 2;                    // 0 -> x0/sim1, 1 -> x1/sim2
  const int b     = z & 3;

  const __bf16* A  = kbf + (size_t)h * LL * LL;                 // L x L row-major
  const __bf16* Bt = (which ? x1t : x0t) + (size_t)b * DD * LL; // D x L (= B^T)
  __bf16* outp     = (which ? sim2 : sim1);
  const int bh     = b * HH + h;

  const v8f vzero = {0.f,0.f,0.f,0.f,0.f,0.f,0.f,0.f};
  v8f acc[4][4];
#pragma unroll
  for (int i = 0; i < 4; ++i)
#pragma unroll
    for (int j = 0; j < 4; ++j) acc[i][j] = vzero;

  for (int k0 = 0; k0 < LL; k0 += 32) {
    v16bf af[4];
#pragma unroll
    for (int i = 0; i < 4; ++i) {
      const __bf16* pa = A + (size_t)(lt * 64 + i * 16 + (lane & 15)) * LL + k0
                           + ((lane & 16) ? 8 : 0);
      // prefetch the HBM-streamed A operand 8 K-steps (512B) ahead
      if (k0 + 256 < LL) __builtin_prefetch(pa + 256, 0, 0);
      v8bf lo = *(const v8bf*)pa;
      v8bf hi = *(const v8bf*)(pa + 16);
      af[i] = __builtin_shufflevector(lo, hi, 0,1,2,3,4,5,6,7,8,9,10,11,12,13,14,15);
    }
#pragma unroll
    for (int j = 0; j < 4; ++j) {
      v16bf bf = frag_b(Bt, LL, j * 16 + (lane & 15), k0, lane);
#pragma unroll
      for (int i = 0; i < 4; ++i) acc[i][j] = wmma_bf(af[i], bf, acc[i][j]);
    }
  }

#pragma unroll
  for (int i = 0; i < 4; ++i)
#pragma unroll
    for (int j = 0; j < 4; ++j)
#pragma unroll
      for (int r = 0; r < 8; ++r) {
        int M = lt * 64 + i * 16 + r + ((lane & 16) ? 8 : 0);
        int N = j * 16 + (lane & 15);
        outp[((size_t)bh * LL + M) * DD + N] = (__bf16)acc[i][j][r];
      }
}

// ---------------------------------------------------------------------------
// Column-sum pass: per wave one 64x64 scores tile; scores = sim1 @ sim2^T,
// mask (col>row -> 0), exp, reduce columns, write deterministic partials
// colpart[(bh*32 + lt)*L + m].  grid (256, H, B), block 128.
// ---------------------------------------------------------------------------
__global__ __launch_bounds__(128, 1)
void k_colsum(const __bf16* __restrict__ sim1,
              const __bf16* __restrict__ sim2,
              float* __restrict__ colpart) {
  const int lane = threadIdx.x & 31;
  const int wave = threadIdx.x >> 5;
  const int t  = blockIdx.x * 4 + wave;  // 0..1023
  const int lt = t >> 5;                 // M tile
  const int mt = t & 31;                 // N tile
  const int h  = blockIdx.y;
  const int b  = blockIdx.z;
  const int bh = b * HH + h;

  const __bf16* A  = sim1 + (size_t)bh * LL * DD;  // L x D
  const __bf16* Bt = sim2 + (size_t)bh * LL * DD;  // (scores B)^T = sim2 row-major

  const v8f vzero = {0.f,0.f,0.f,0.f,0.f,0.f,0.f,0.f};
  v8f acc[4][4];
#pragma unroll
  for (int i = 0; i < 4; ++i)
#pragma unroll
    for (int j = 0; j < 4; ++j) acc[i][j] = vzero;

#pragma unroll
  for (int ks = 0; ks < 2; ++ks) {
    int k0 = ks * 32;
    v16bf af[4];
#pragma unroll
    for (int i = 0; i < 4; ++i)
      af[i] = frag_a(A, DD, lt * 64 + i * 16 + (lane & 15), k0, lane);
#pragma unroll
    for (int j = 0; j < 4; ++j) {
      v16bf bf = frag_b(Bt, DD, mt * 64 + j * 16 + (lane & 15), k0, lane);
#pragma unroll
      for (int i = 0; i < 4; ++i) acc[i][j] = wmma_bf(af[i], bf, acc[i][j]);
    }
  }

  float csum[4] = {0.f, 0.f, 0.f, 0.f};
#pragma unroll
  for (int j = 0; j < 4; ++j) {
    int N = mt * 64 + j * 16 + (lane & 15);
#pragma unroll
    for (int i = 0; i < 4; ++i)
#pragma unroll
      for (int r = 0; r < 8; ++r) {
        int M = lt * 64 + i * 16 + r + ((lane & 16) ? 8 : 0);
        float p = (N > M) ? 0.f : __expf(acc[i][j][r]);
        csum[j] += p;
      }
    csum[j] += __shfl_xor(csum[j], 16, 32);  // fold lanes 16-31 (M+8 rows)
  }
  if (lane < 16) {
#pragma unroll
    for (int j = 0; j < 4; ++j)
      colpart[((size_t)bh * 32 + lt) * LL + mt * 64 + j * 16 + lane] = csum[j];
  }
}

// reduce partials over 32 l-tiles (fixed order => deterministic), store 1/sum
__global__ __launch_bounds__(256) void k_colreduce(const float* __restrict__ colpart,
                                                   float* __restrict__ rcol) {
  int i = blockIdx.x * blockDim.x + threadIdx.x;  // bh*L + m, n = B*H*L
  if (i >= BB * HH * LL) return;
  int bh = i / LL, m = i - bh * LL;
  float s = 0.f;
#pragma unroll
  for (int t = 0; t < 32; ++t) s += colpart[((size_t)bh * 32 + t) * LL + m];
  rcol[i] = 1.0f / s;
}

// ---------------------------------------------------------------------------
// heads: per wave one 64(l) x 64(d) output tile, streaming over 32 m-tiles.
// Recompute scores tile, p = exp(s)*rcol (masked), LDS round-trip to A-layout,
// WMMA against x1^T.  grid (8, H, B), block 128, 32KB LDS.
// ---------------------------------------------------------------------------
__global__ __launch_bounds__(128, 1)
void k_heads(const __bf16* __restrict__ sim1,
             const __bf16* __restrict__ sim2,
             const __bf16* __restrict__ x1t,
             const float* __restrict__ rcol,
             float* __restrict__ heads) {
  __shared__ __bf16 pbuf[4 * 64 * 64];  // 8KB per wave
  const int lane = threadIdx.x & 31;
  const int wave = threadIdx.x >> 5;
  const int lt = blockIdx.x * 4 + wave;  // 0..31
  const int h  = blockIdx.y;
  const int b  = blockIdx.z;
  const int bh = b * HH + h;

  const __bf16* A1 = sim1 + (size_t)bh * LL * DD;
  const __bf16* A2 = sim2 + (size_t)bh * LL * DD;
  const __bf16* Xt = x1t + (size_t)b * DD * LL;   // D x L
  __bf16* pw = pbuf + wave * 64 * 64;

  const v8f vzero = {0.f,0.f,0.f,0.f,0.f,0.f,0.f,0.f};
  v8f hc[4][4];
#pragma unroll
  for (int i = 0; i < 4; ++i)
#pragma unroll
    for (int j = 0; j < 4; ++j) hc[i][j] = vzero;

  for (int mt = 0; mt < 32; ++mt) {
    // ---- build p tile (64 l x 64 m) in LDS as bf16 ----
#pragma unroll
    for (int j = 0; j < 4; ++j) {
      v8f sc[4];
#pragma unroll
      for (int i = 0; i < 4; ++i) sc[i] = vzero;
#pragma unroll
      for (int ks = 0; ks < 2; ++ks) {
        int k0 = ks * 32;
        v16bf bf = frag_b(A2, DD, mt * 64 + j * 16 + (lane & 15), k0, lane);
#pragma unroll
        for (int i = 0; i < 4; ++i) {
          v16bf af = frag_a(A1, DD, lt * 64 + i * 16 + (lane & 15), k0, lane);
          sc[i] = wmma_bf(af, bf, sc[i]);
        }
      }
      int N = mt * 64 + j * 16 + (lane & 15);
      float rsv = rcol[(size_t)bh * LL + N];
#pragma unroll
      for (int i = 0; i < 4; ++i)
#pragma unroll
        for (int r = 0; r < 8; ++r) {
          int Mloc = i * 16 + r + ((lane & 16) ? 8 : 0);
          int Mg = lt * 64 + Mloc;
          float p = (N > Mg) ? 0.f : __expf(sc[i][r]) * rsv;
          pw[Mloc * 64 + j * 16 + (lane & 15)] = (__bf16)p;
        }
    }
    __syncthreads();

    // ---- heads += p (A from LDS) @ x1 (B from x1^T) ----
#pragma unroll
    for (int ks = 0; ks < 2; ++ks) {
      int kp = ks * 32;
      v16bf ap[4];
#pragma unroll
      for (int i = 0; i < 4; ++i)
        ap[i] = frag_a(pw, 64, i * 16 + (lane & 15), kp, lane);
#pragma unroll
      for (int jd = 0; jd < 4; ++jd) {
        v16bf bf = frag_b(Xt, LL, jd * 16 + (lane & 15), mt * 64 + kp, lane);
#pragma unroll
        for (int i = 0; i < 4; ++i) hc[i][jd] = wmma_bf(ap[i], bf, hc[i][jd]);
      }
    }
    __syncthreads();
  }

#pragma unroll
  for (int i = 0; i < 4; ++i)
#pragma unroll
    for (int jd = 0; jd < 4; ++jd)
#pragma unroll
      for (int r = 0; r < 8; ++r) {
        int M = lt * 64 + i * 16 + r + ((lane & 16) ? 8 : 0);
        int N = jd * 16 + (lane & 15);
        heads[((size_t)bh * LL + M) * DD + N] = hc[i][jd][r];
      }
}

// out[b,l,d] = conv_b + sum_h conv_w[h] * heads[b,h,l,d]
__global__ __launch_bounds__(256) void k_combine(const float* __restrict__ heads,
                                                 const float* __restrict__ w,
                                                 const float* __restrict__ bias,
                                                 float* __restrict__ out) {
  const int n = BB * LL * DD;
  for (int i = blockIdx.x * blockDim.x + threadIdx.x; i < n; i += gridDim.x * blockDim.x) {
    int b = i / (LL * DD);
    int ld = i - b * (LL * DD);
    float s = bias[0];
#pragma unroll
    for (int h = 0; h < HH; ++h)
      s += w[h] * heads[((size_t)(b * HH + h) * LL * DD) + ld];
    out[i] = s;
  }
}

// ---------------------------------------------------------------------------
extern "C" void kernel_launch(void* const* d_in, const int* in_sizes, int n_in,
                              void* d_out, int out_size, void* d_ws, size_t ws_size,
                              hipStream_t stream) {
  const float* x0   = (const float*)d_in[0];
  const float* x1   = (const float*)d_in[1];
  const float* kerf = (const float*)d_in[2];
  const float* cw   = (const float*)d_in[3];
  const float* cb   = (const float*)d_in[4];
  float* out = (float*)d_out;

  char* ws = (char*)d_ws;
  size_t off = 0;
  auto take = [&](size_t bytes) { char* p = ws + off; off = (off + bytes + 255) & ~(size_t)255; return p; };

  __bf16* kbf     = (__bf16*)take((size_t)HH * LL * LL * 2);        // 64 MB
  __bf16* x0t     = (__bf16*)take((size_t)BB * DD * LL * 2);        // 1 MB
  __bf16* x1t     = (__bf16*)take((size_t)BB * DD * LL * 2);        // 1 MB
  __bf16* sim1    = (__bf16*)take((size_t)BB * HH * LL * DD * 2);   // 8 MB
  __bf16* sim2    = (__bf16*)take((size_t)BB * HH * LL * DD * 2);   // 8 MB
  float*  colpart = (float*) take((size_t)BB * HH * 32 * LL * 4);   // 8 MB
  float*  rcol    = (float*) take((size_t)BB * HH * LL * 4);        // 256 KB
  float*  heads   = (float*) take((size_t)BB * HH * LL * DD * 4);   // 16 MB

  k_f32_to_bf16<<<2048, 256, 0, stream>>>(kerf, kbf, HH * LL * LL);
  k_transpose<<<1024, 256, 0, stream>>>(x0, x0t);
  k_transpose<<<1024, 256, 0, stream>>>(x1, x1t);
  k_gemm1<<<dim3(8, HH, 2 * BB), 128, 0, stream>>>(kbf, x0t, x1t, sim1, sim2);
  k_colsum<<<dim3(256, HH, BB), 128, 0, stream>>>(sim1, sim2, colpart);
  k_colreduce<<<256, 256, 0, stream>>>(colpart, rcol);
  k_heads<<<dim3(8, HH, BB), 128, 0, stream>>>(sim1, sim2, x1t, rcol, heads);
  k_combine<<<2048, 256, 0, stream>>>(heads, cw, cb, out);
}
